// E_GCL_75359496176069
// MI455X (gfx1250) — matrix operand order repaired and checked by reading, not compiled, not run on hardware
//
#include <hip/hip_runtime.h>
#include <hip/hip_bf16.h>

typedef __attribute__((ext_vector_type(16))) _Float16 v16h;
typedef __attribute__((ext_vector_type(8)))  _Float16 v8h;
typedef __attribute__((ext_vector_type(8)))  float    v8f;
typedef __attribute__((ext_vector_type(4)))  float    v4f;

#define BLK 512                  // 16 waves: 4 waves/SIMD even at 1 block/WGP
#define WPB 16                   // waves per block
#define SCR_STRIDE 136           // 128 + 8 halves pad, 272B rows (16B aligned)
#define SCR_PER_WAVE (16 * SCR_STRIDE)

__device__ __forceinline__ float silu_f(float x) {
    return x / (1.0f + __expf(-x));
}

__device__ __forceinline__ v8f wmma_f16(v16h a, v16h b, v8f c) {
    // D = A(16x32 f16) * B(32x16 f16) + C(16x16 f32)
    return __builtin_amdgcn_wmma_f32_16x16x32_f16(
        /*neg_a=*/false, a, /*neg_b=*/false, b,
        /*c_mod=*/(short)0, c, /*reuse_a=*/false, /*reuse_b=*/false);
}

// A-matrix 16x32 f16 layout (ISA 7.12.2): lane holds row M=lane&15,
// K = kbase..kbase+7 and kbase+16..kbase+23, kbase = (lane>>4)*8.
__device__ __forceinline__ v16h load_A_h(const _Float16* rowp, int k0, int lh) {
    const int kb = k0 + lh * 8;
    union { v16h v; v8h h[2]; } u;
    u.h[0] = *(const v8h*)(rowp + kb);
    u.h[1] = *(const v8h*)(rowp + kb + 16);
    return u.v;
}

// Same A layout, f32 source rows (convert on load).
__device__ __forceinline__ v16h load_A_f(const float* rowp, int k0, int lh) {
    const int kb = k0 + lh * 8;
    v4f a0 = *(const v4f*)(rowp + kb);
    v4f a1 = *(const v4f*)(rowp + kb + 4);
    v4f b0 = *(const v4f*)(rowp + kb + 16);
    v4f b1 = *(const v4f*)(rowp + kb + 20);
    v16h r;
#pragma unroll
    for (int i = 0; i < 4; i++) {
        r[i]      = (_Float16)a0[i];
        r[4 + i]  = (_Float16)a1[i];
        r[8 + i]  = (_Float16)b0[i];
        r[12 + i] = (_Float16)b1[i];
    }
    return r;
}

// B-matrix 32x16 f16 layout: lane holds column N = n0 + (lane&15),
// K = kb..kb+7 and kb+16..kb+23, kb = k0 + (lane>>4)*8 (WT stored [N][K]).
__device__ __forceinline__ v16h load_B_h(const _Float16* WT, int KP, int n0, int k0, int lane) {
    const _Float16* p = WT + (size_t)(n0 + (lane & 15)) * KP + k0 + ((lane >> 4) << 3);
    union { v16h v; v8h h[2]; } u;
    u.h[0] = *(const v8h*)p;
    u.h[1] = *(const v8h*)(p + 16);
    return u.v;
}

__device__ __forceinline__ void lds_copy16(void* dst, const void* src, int bytes, int tid, int nthr) {
    int4* d = (int4*)dst;
    const int4* s = (const int4*)src;
    for (int i = tid; i < bytes / 16; i += nthr) d[i] = s[i];
}

// ---------------- prep kernels ----------------

__global__ void cvt_h_kernel(const float* __restrict__ x, _Float16* __restrict__ y, int n) {
    int i = blockIdx.x * 256 + threadIdx.x;
    if (i < n) y[i] = (_Float16)x[i];
}

// W row-major [K][Ncols] f32 -> WT [Ncols][K] f16
__global__ void transpose_cvt_kernel(const float* __restrict__ W, _Float16* __restrict__ WT,
                                     int K, int Ncols) {
    int i = blockIdx.x * 256 + threadIdx.x;
    if (i < K * Ncols) {
        int k = i / Ncols, n = i % Ncols;
        WT[(size_t)n * K + k] = (_Float16)W[i];
    }
}

// ---------------- fused edge kernel ----------------
// block = 512 (16 waves), each wave processes 16 edges through
// edge_mlp (2 GEMMs) + coord_mlp (1 GEMM + dot), scattering segment sums.
// N-tiles processed in pairs with two accumulators so consecutive WMMAs
// are independent (avoids the 5-slot f16 WMMA->WMMA RAW hazard).

__global__ __launch_bounds__(BLK) void egcl_edge_kernel(
    const _Float16* __restrict__ h16, const float* __restrict__ coord,
    const int* __restrict__ rowi, const int* __restrict__ coli,
    const _Float16* __restrict__ We1T, const float* __restrict__ We1r,
    const float* __restrict__ be1,
    const _Float16* __restrict__ We2T, const float* __restrict__ be2,
    const _Float16* __restrict__ Wc1T, const float* __restrict__ bc1,
    const float* __restrict__ Wc2,
    float* __restrict__ agg_h, float* __restrict__ agg_c,
    float* __restrict__ cnt, int nE)
{
    extern __shared__ char smem[];
    _Float16* sWe1T = (_Float16*)smem;            // 256*128 halves
    _Float16* sWe2T = sWe1T + 256 * 128;          // 128*128
    _Float16* sWc1T = sWe2T + 128 * 128;          // 128*128
    float* sWe1r = (float*)(sWc1T + 128 * 128);   // 128 (radial row of We1, f32)
    float* sbe1  = sWe1r + 128;
    float* sbe2  = sbe1 + 128;
    float* sbc1  = sbe2 + 128;
    float* sWc2  = sbc1 + 128;
    float* sRad  = sWc2 + 128;                    // WPB * 16
    float* sPhi  = sRad + WPB * 16;               // WPB * 16
    int*   sRow  = (int*)(sPhi + WPB * 16);       // WPB * 16
    _Float16* sScrAll = (_Float16*)(sRow + WPB * 16); // WPB * 16*136 halves

    const int tid = threadIdx.x;
    lds_copy16(sWe1T, We1T, 256 * 128 * 2, tid, BLK);
    lds_copy16(sWe2T, We2T, 128 * 128 * 2, tid, BLK);
    lds_copy16(sWc1T, Wc1T, 128 * 128 * 2, tid, BLK);
    if (tid < 128) {
        sWe1r[tid] = We1r[tid];
        sbe1[tid]  = be1[tid];
        sbe2[tid]  = be2[tid];
        sbc1[tid]  = bc1[tid];
        sWc2[tid]  = Wc2[tid];
    }
    __syncthreads();

    const int wave = tid >> 5, lane = tid & 31;
    const int lm = lane & 15, lh = lane >> 4;
    const int tile = blockIdx.x * WPB + wave;
    const int e0 = tile * 16;
    if (e0 >= nE) return;                         // wave-uniform
    _Float16* scr = sScrAll + wave * SCR_PER_WAVE;

    const int e  = e0 + lm;                       // both half-waves read same edge
    const int rn = rowi[e];
    const int cn = coli[e];
    float dx = 0.f, dy = 0.f, dz = 0.f;
    if (lh == 0) {
        dx = coord[rn * 3 + 0] - coord[cn * 3 + 0];
        dy = coord[rn * 3 + 1] - coord[cn * 3 + 1];
        dz = coord[rn * 3 + 2] - coord[cn * 3 + 2];
        sRad[wave * 16 + lm] = dx * dx + dy * dy + dz * dz;
        sRow[wave * 16 + lm] = rn;
    }

    // ---- A tile for GEMM1: e_in[:,0:256] = [h[row] | h[col]] ----
    v16h A[8];
    const _Float16* hr = h16 + (size_t)rn * 128;
    const _Float16* hc = h16 + (size_t)cn * 128;
#pragma unroll
    for (int c = 0; c < 4; c++) {
        A[c]     = load_A_h(hr, c * 32, lh);
        A[4 + c] = load_A_h(hc, c * 32, lh);
    }
    asm volatile("s_wait_dscnt 0x0" ::: "memory");  // sRad/sRow visible (same wave)

    // ---- GEMM1: e_in @ We1 + be1 + radial*We1[256,:] -> SiLU -> scratch ----
#pragma unroll
    for (int nt = 0; nt < 4; nt++) {
        const int n0 = nt * 32;
        v8f acc0 = {}, acc1 = {};
#pragma unroll
        for (int c = 0; c < 8; c++) {
            v16h b0 = load_B_h(sWe1T, 256, n0,      c * 32, lane);
            v16h b1 = load_B_h(sWe1T, 256, n0 + 16, c * 32, lane);
            acc0 = wmma_f16(A[c], b0, acc0);      // independent pair: no RAW NOPs
            acc1 = wmma_f16(A[c], b1, acc1);
        }
        const int nA = n0 + lm, nB = n0 + 16 + lm;
        const float biasA = sbe1[nA], w1rA = sWe1r[nA];
        const float biasB = sbe1[nB], w1rB = sWe1r[nB];
#pragma unroll
        for (int r = 0; r < 8; r++) {
            const int M = lh * 8 + r;             // D layout: M = (lane>=16?8:0)+r
            const float rad = sRad[wave * 16 + M];
            scr[M * SCR_STRIDE + nA] = (_Float16)silu_f(acc0[r] + biasA + rad * w1rA);
            scr[M * SCR_STRIDE + nB] = (_Float16)silu_f(acc1[r] + biasB + rad * w1rB);
        }
    }
    asm volatile("s_wait_dscnt 0x0" ::: "memory");

    // ---- GEMM2: act1 @ We2 + be2 -> SiLU = edge_feat; scatter + scratch ----
    v16h A2[4];
#pragma unroll
    for (int c = 0; c < 4; c++) A2[c] = load_A_h(scr + lm * SCR_STRIDE, c * 32, lh);
    asm volatile("s_wait_dscnt 0x0" ::: "memory");    // loads done before overwrite
#pragma unroll
    for (int nt = 0; nt < 4; nt++) {
        const int n0 = nt * 32;
        v8f acc0 = {}, acc1 = {};
#pragma unroll
        for (int c = 0; c < 4; c++) {
            v16h b0 = load_B_h(sWe2T, 128, n0,      c * 32, lane);
            v16h b1 = load_B_h(sWe2T, 128, n0 + 16, c * 32, lane);
            acc0 = wmma_f16(A2[c], b0, acc0);
            acc1 = wmma_f16(A2[c], b1, acc1);
        }
        const int nA = n0 + lm, nB = n0 + 16 + lm;
        const float bA = sbe2[nA], bB = sbe2[nB];
#pragma unroll
        for (int r = 0; r < 8; r++) {
            const int M = lh * 8 + r;
            const size_t rbase = (size_t)sRow[wave * 16 + M] * 128;
            float efA = silu_f(acc0[r] + bA);
            float efB = silu_f(acc1[r] + bB);
            atomicAdd(&agg_h[rbase + nA], efA);
            atomicAdd(&agg_h[rbase + nB], efB);
            scr[M * SCR_STRIDE + nA] = (_Float16)efA;
            scr[M * SCR_STRIDE + nB] = (_Float16)efB;
        }
    }
    asm volatile("s_wait_dscnt 0x0" ::: "memory");

    // ---- GEMM3: silu(edge_feat @ Wc1 + bc1) . Wc2 -> phi ----
    v16h A3[4];
#pragma unroll
    for (int c = 0; c < 4; c++) A3[c] = load_A_h(scr + lm * SCR_STRIDE, c * 32, lh);
    float pphi[8];
#pragma unroll
    for (int r = 0; r < 8; r++) pphi[r] = 0.f;
#pragma unroll
    for (int nt = 0; nt < 4; nt++) {
        const int n0 = nt * 32;
        v8f acc0 = {}, acc1 = {};
#pragma unroll
        for (int c = 0; c < 4; c++) {
            v16h b0 = load_B_h(sWc1T, 128, n0,      c * 32, lane);
            v16h b1 = load_B_h(sWc1T, 128, n0 + 16, c * 32, lane);
            acc0 = wmma_f16(A3[c], b0, acc0);
            acc1 = wmma_f16(A3[c], b1, acc1);
        }
        const int nA = n0 + lm, nB = n0 + 16 + lm;
        const float bcA = sbc1[nA], wA = sWc2[nA];
        const float bcB = sbc1[nB], wB = sWc2[nB];
#pragma unroll
        for (int r = 0; r < 8; r++) {
            pphi[r] += silu_f(acc0[r] + bcA) * wA;
            pphi[r] += silu_f(acc1[r] + bcB) * wB;
        }
    }
    // reduce over the 16 N-lanes within each half-wave (xor 1,2,4,8 stays in group)
#pragma unroll
    for (int off = 1; off < 16; off <<= 1) {
#pragma unroll
        for (int r = 0; r < 8; r++) pphi[r] += __shfl_xor(pphi[r], off, 32);
    }
    if (lm == 0) {
#pragma unroll
        for (int r = 0; r < 8; r++) sPhi[wave * 16 + lh * 8 + r] = pphi[r];
    }
    asm volatile("s_wait_dscnt 0x0" ::: "memory");

    // ---- coord scatter (lanes 0-15, one edge each) ----
    if (lh == 0) {
        const float phi = sPhi[wave * 16 + lm];
        atomicAdd(&agg_c[rn * 3 + 0], dx * phi);
        atomicAdd(&agg_c[rn * 3 + 1], dy * phi);
        atomicAdd(&agg_c[rn * 3 + 2], dz * phi);
        atomicAdd(&cnt[rn], 1.0f);
    }
}

// ---------------- node kernel ----------------
// h_out = h + silu(concat(h, agg_h) @ Wn1 + bn1) @ Wn2 + bn2

__global__ __launch_bounds__(BLK) void egcl_node_kernel(
    const float* __restrict__ h, const _Float16* __restrict__ h16,
    const float* __restrict__ agg_h,
    const _Float16* __restrict__ Wn1T, const float* __restrict__ bn1,
    const _Float16* __restrict__ Wn2T, const float* __restrict__ bn2,
    float* __restrict__ h_out, int nN)
{
    extern __shared__ char smem[];
    _Float16* sWn1T = (_Float16*)smem;            // 256*128
    _Float16* sWn2T = sWn1T + 256 * 128;          // 128*128
    float* sbn1 = (float*)(sWn2T + 128 * 128);
    float* sbn2 = sbn1 + 128;
    _Float16* sScrAll = (_Float16*)(sbn2 + 128);

    const int tid = threadIdx.x;
    lds_copy16(sWn1T, Wn1T, 256 * 128 * 2, tid, BLK);
    lds_copy16(sWn2T, Wn2T, 128 * 128 * 2, tid, BLK);
    if (tid < 128) { sbn1[tid] = bn1[tid]; sbn2[tid] = bn2[tid]; }
    __syncthreads();

    const int wave = tid >> 5, lane = tid & 31;
    const int lm = lane & 15, lh = lane >> 4;
    const int ntiles = nN / 16;                   // N is a multiple of 16
    int tile = blockIdx.x * WPB + wave;
    if (tile >= ntiles) tile = ntiles - 1;        // uniform clamp: duplicate identical work
    const int node0 = tile * 16;
    _Float16* scr = sScrAll + wave * SCR_PER_WAVE;

    const int node = node0 + lm;                  // A row for this lane
    v16h A[8];
    const _Float16* hn = h16 + (size_t)node * 128;
    const float* an = agg_h + (size_t)node * 128;
#pragma unroll
    for (int c = 0; c < 4; c++) {
        A[c]     = load_A_h(hn, c * 32, lh);      // K 0..127  = h
        A[4 + c] = load_A_f(an, c * 32, lh);      // K 128..255 = agg_h
    }

    // GEMM1 + bias + SiLU -> scratch
#pragma unroll
    for (int nt = 0; nt < 4; nt++) {
        const int n0 = nt * 32;
        v8f acc0 = {}, acc1 = {};
#pragma unroll
        for (int c = 0; c < 8; c++) {
            v16h b0 = load_B_h(sWn1T, 256, n0,      c * 32, lane);
            v16h b1 = load_B_h(sWn1T, 256, n0 + 16, c * 32, lane);
            acc0 = wmma_f16(A[c], b0, acc0);
            acc1 = wmma_f16(A[c], b1, acc1);
        }
        const int nA = n0 + lm, nB = n0 + 16 + lm;
        const float bA = sbn1[nA], bB = sbn1[nB];
#pragma unroll
        for (int r = 0; r < 8; r++) {
            const int M = lh * 8 + r;
            scr[M * SCR_STRIDE + nA] = (_Float16)silu_f(acc0[r] + bA);
            scr[M * SCR_STRIDE + nB] = (_Float16)silu_f(acc1[r] + bB);
        }
    }
    asm volatile("s_wait_dscnt 0x0" ::: "memory");

    // GEMM2 + bias + residual -> h_out
    v16h A2[4];
#pragma unroll
    for (int c = 0; c < 4; c++) A2[c] = load_A_h(scr + lm * SCR_STRIDE, c * 32, lh);
#pragma unroll
    for (int nt = 0; nt < 4; nt++) {
        const int n0 = nt * 32;
        v8f acc0 = {}, acc1 = {};
#pragma unroll
        for (int c = 0; c < 4; c++) {
            v16h b0 = load_B_h(sWn2T, 128, n0,      c * 32, lane);
            v16h b1 = load_B_h(sWn2T, 128, n0 + 16, c * 32, lane);
            acc0 = wmma_f16(A2[c], b0, acc0);
            acc1 = wmma_f16(A2[c], b1, acc1);
        }
        const int nA = n0 + lm, nB = n0 + 16 + lm;
        const float bA = sbn2[nA], bB = sbn2[nB];
#pragma unroll
        for (int r = 0; r < 8; r++) {
            const int nodeM = node0 + lh * 8 + r;
            const size_t offA = (size_t)nodeM * 128 + nA;
            const size_t offB = (size_t)nodeM * 128 + nB;
            h_out[offA] = h[offA] + acc0[r] + bA;
            h_out[offB] = h[offB] + acc1[r] + bB;
        }
    }
}

__global__ void coord_out_kernel(const float* __restrict__ coord,
                                 const float* __restrict__ agg_c,
                                 const float* __restrict__ cnt,
                                 float* __restrict__ out, int nN) {
    int i = blockIdx.x * 256 + threadIdx.x;
    if (i < nN * 3) {
        int node = i / 3;
        out[i] = coord[i] + agg_c[i] / fmaxf(cnt[node], 1.0f);
    }
}

// ---------------- launch ----------------

static inline size_t alignup(size_t x) { return (x + 255) & ~(size_t)255; }

extern "C" void kernel_launch(void* const* d_in, const int* in_sizes, int n_in,
                              void* d_out, int out_size, void* d_ws, size_t ws_size,
                              hipStream_t stream) {
    const float* h     = (const float*)d_in[0];
    const float* coord = (const float*)d_in[1];
    const int*   eidx  = (const int*)  d_in[2];
    const float* We1   = (const float*)d_in[3];
    const float* be1   = (const float*)d_in[4];
    const float* We2   = (const float*)d_in[5];
    const float* be2   = (const float*)d_in[6];
    const float* Wn1   = (const float*)d_in[7];
    const float* bn1   = (const float*)d_in[8];
    const float* Wn2   = (const float*)d_in[9];
    const float* bn2   = (const float*)d_in[10];
    const float* Wc1   = (const float*)d_in[11];
    const float* bc1   = (const float*)d_in[12];
    const float* Wc2   = (const float*)d_in[13];

    const int F = 128, H = 128;
    const int N = in_sizes[0] / F;
    const int E = in_sizes[2] / 2;
    const int* rowi = eidx;
    const int* coli = eidx + E;

    char* p = (char*)d_ws;
    _Float16* h16  = (_Float16*)p;  p += alignup((size_t)N * F * 2);
    float* agg_h   = (float*)p;     p += alignup((size_t)N * H * 4);
    float* agg_c   = (float*)p;     p += alignup((size_t)N * 3 * 4);
    float* cnt     = (float*)p;     p += alignup((size_t)N * 4);
    _Float16* We1T = (_Float16*)p;  p += alignup((size_t)256 * 128 * 2);
    _Float16* We2T = (_Float16*)p;  p += alignup((size_t)128 * 128 * 2);
    _Float16* Wc1T = (_Float16*)p;  p += alignup((size_t)128 * 128 * 2);
    _Float16* Wn1T = (_Float16*)p;  p += alignup((size_t)256 * 128 * 2);
    _Float16* Wn2T = (_Float16*)p;  p += alignup((size_t)128 * 128 * 2);
    (void)ws_size; (void)n_in;

    hipMemsetAsync(agg_h, 0, (size_t)N * H * 4, stream);
    hipMemsetAsync(agg_c, 0, (size_t)N * 3 * 4, stream);
    hipMemsetAsync(cnt,   0, (size_t)N * 4,     stream);

    cvt_h_kernel<<<(N * F + 255) / 256, 256, 0, stream>>>(h, h16, N * F);
    transpose_cvt_kernel<<<(256 * 128 + 255) / 256, 256, 0, stream>>>(We1, We1T, 256, 128);
    transpose_cvt_kernel<<<(128 * 128 + 255) / 256, 256, 0, stream>>>(We2, We2T, 128, 128);
    transpose_cvt_kernel<<<(128 * 128 + 255) / 256, 256, 0, stream>>>(Wc1, Wc1T, 128, 128);
    transpose_cvt_kernel<<<(256 * 128 + 255) / 256, 256, 0, stream>>>(Wn1, Wn1T, 256, 128);
    transpose_cvt_kernel<<<(128 * 128 + 255) / 256, 256, 0, stream>>>(Wn2, Wn2T, 128, 128);

    const size_t edge_lds =
        (size_t)(256 * 128 + 128 * 128 + 128 * 128) * 2   // f16 weights
        + 5 * 128 * 4                                     // We1r, be1, be2, bc1, Wc2
        + 2 * WPB * 16 * 4                                // sRad, sPhi
        + WPB * 16 * 4                                    // sRow (int)
        + (size_t)WPB * SCR_PER_WAVE * 2;                 // scratch tiles
    egcl_edge_kernel<<<(E + WPB * 16 - 1) / (WPB * 16), BLK, edge_lds, stream>>>(
        h16, coord, rowi, coli,
        We1T, We1 + 256 * 128, be1, We2T, be2, Wc1T, bc1, Wc2,
        agg_h, agg_c, cnt, E);

    float* h_out = (float*)d_out;
    float* c_out = h_out + (size_t)N * F;
    const size_t node_lds =
        (size_t)(256 * 128 + 128 * 128) * 2
        + 2 * 128 * 4
        + (size_t)WPB * SCR_PER_WAVE * 2;
    egcl_node_kernel<<<(N + WPB * 16 - 1) / (WPB * 16), BLK, node_lds, stream>>>(
        h, h16, agg_h, Wn1T, bn1, Wn2T, bn2, h_out, N);

    coord_out_kernel<<<(N * 3 + 255) / 256, 256, 0, stream>>>(coord, agg_c, cnt, c_out, N);

    (void)out_size;
}
